// ToMeBlock_69982197121294
// MI455X (gfx1250) — matrix operand order repaired and checked by reading, not compiled, not run on hardware
//
#include <hip/hip_runtime.h>
#include <hip/hip_bf16.h>
#include <cstdint>

// ---------------- problem constants ----------------
#define BB   32
#define NN   577
#define DD   1024
#define HH   16
#define RR   16
#define HDD  64
#define HIDW 4096
#define NP   576          // N-1 patches
#define NSRC 288
#define NDST 288
#define MM   561          // N - R merged tokens
#define TOK1 (BB*NN)      // 18464
#define TOK2 (BB*MM)      // 17952

typedef __attribute__((ext_vector_type(16))) __bf16 v16bf;
typedef __attribute__((ext_vector_type(8)))  float  v8f;

union FragBF { uint4 u4[2]; v16bf v; };

__device__ __forceinline__ unsigned short f2bf(float f) {
  unsigned int u = __float_as_uint(f);
  unsigned int r = u + 0x7FFFu + ((u >> 16) & 1u);   // round-to-nearest-even
  return (unsigned short)(r >> 16);
}
__device__ __forceinline__ v8f v8f_zero() {
  v8f z = {0.f,0.f,0.f,0.f,0.f,0.f,0.f,0.f}; return z;
}
__device__ __forceinline__ float gelu_exact(float v) {
  return 0.5f * v * (1.0f + erff(v * 0.70710678118654752440f));
}
__device__ __forceinline__ float rmax16(float v) {
  #pragma unroll
  for (int m = 1; m < 16; m <<= 1) v = fmaxf(v, __shfl_xor(v, m, 32));
  return v;
}
__device__ __forceinline__ float rsum16(float v) {
  #pragma unroll
  for (int m = 1; m < 16; m <<= 1) v += __shfl_xor(v, m, 32);
  return v;
}

// ---------------- elementwise f32 -> bf16 ----------------
__global__ void cvt_bf16(const float* __restrict__ in, unsigned short* __restrict__ out, int n) {
  int i = blockIdx.x * 256 + threadIdx.x;
  if (i < n) out[i] = f2bf(in[i]);
}

// ---------------- head-averaged key weights ----------------
__global__ void avg_k_weights(const float* __restrict__ qkv_w, const float* __restrict__ qkv_b,
                              unsigned short* __restrict__ wk, float* __restrict__ bk) {
  int idx = blockIdx.x * 256 + threadIdx.x;
  if (idx < DD * HDD) {
    int hd = idx & (HDD - 1), d = idx >> 6;
    float s = 0.f;
    #pragma unroll
    for (int h = 0; h < HH; ++h) s += qkv_w[(size_t)d * (3*DD) + DD + h*HDD + hd];
    wk[d*HDD + hd] = f2bf(s * (1.f/HH));
  }
  if (idx < HDD) {
    float s = 0.f;
    #pragma unroll
    for (int h = 0; h < HH; ++h) s += qkv_b[DD + h*HDD + idx];
    bk[idx] = s * (1.f/HH);
  }
}

// ---------------- LayerNorm over D=1024 -> bf16 (one block per row) ----------------
__global__ __launch_bounds__(256)
void layernorm_bf16(const float* __restrict__ X, const float* __restrict__ g,
                    const float* __restrict__ bta, unsigned short* __restrict__ Y) {
  __shared__ float sb[256], sb2[256];
  const int row = blockIdx.x, tid = threadIdx.x;
  const float* x = X + (size_t)row * DD;
  float v[4], s = 0.f, s2 = 0.f;
  #pragma unroll
  for (int i = 0; i < 4; ++i) { v[i] = x[tid + i*256]; s += v[i]; s2 += v[i]*v[i]; }
  sb[tid] = s; sb2[tid] = s2; __syncthreads();
  for (int st = 128; st > 0; st >>= 1) {
    if (tid < st) { sb[tid] += sb[tid+st]; sb2[tid] += sb2[tid+st]; }
    __syncthreads();
  }
  float mean = sb[0] * (1.f/DD);
  float var  = sb2[0] * (1.f/DD) - mean*mean;
  float inv  = rsqrtf(var + 1e-6f);
  unsigned short* y = Y + (size_t)row * DD;
  #pragma unroll
  for (int i = 0; i < 4; ++i) {
    int c = tid + i*256;
    y[c] = f2bf((v[i] - mean) * inv * g[c] + bta[c]);
  }
}

// ---------------- tile loader for the 128x128 GEMM ----------------
// A tile: 128 rows x 32 k (row-major). B tile: 32 k x 128 n, stored transposed [n][k]
// with (k,k+1) pairs packed so the LDS transpose uses dword stores.
__device__ __forceinline__ void g_load_tiles(const unsigned short* __restrict__ A,
                                             const unsigned short* __restrict__ Bw,
                                             unsigned short* __restrict__ As,
                                             unsigned short* __restrict__ Bs,
                                             int M, int N, int K,
                                             int m0, int n0, int k0, int tid) {
  { // A: thread -> row tid>>1, 16 halves at (tid&1)*16
    int r = tid >> 1, kc = (tid & 1) << 4, gr = m0 + r;
    uint4 d0, d1;
    if (gr < M) {
      const uint4* src = (const uint4*)(A + (size_t)gr * K + k0 + kc);
      d0 = src[0]; d1 = src[1];
    } else { d0 = make_uint4(0,0,0,0); d1 = d0; }
    *(uint4*)&As[r*32 + kc]     = d0;
    *(uint4*)&As[r*32 + kc + 8] = d1;
  }
  { // B: thread -> k-pair (tid&15)*2, 8 cols at (tid>>4)*8
    int kp = (tid & 15) << 1, nc = (tid >> 4) << 3;
    const unsigned short* b0 = Bw + (size_t)(k0 + kp) * N + n0 + nc;
    uint4 d0 = *(const uint4*)b0;
    uint4 d1 = *(const uint4*)(b0 + N);
    const unsigned short* p0 = (const unsigned short*)&d0;
    const unsigned short* p1 = (const unsigned short*)&d1;
    #pragma unroll
    for (int j = 0; j < 8; ++j) {
      unsigned int v = (unsigned int)p0[j] | ((unsigned int)p1[j] << 16);
      *(unsigned int*)&Bs[(nc + j)*32 + kp] = v;
    }
  }
}

// ---------------- main WMMA GEMM: 128x128 block, double-buffered LDS ----------------
// 8 waves as 4(M) x 2(N); each wave: 32 rows x 64 cols = 8 WMMAs / 32-K step.
template<int GELU_, int RESID_, int OUTBF16_>
__global__ __launch_bounds__(256)
void gemm_bf16(const unsigned short* __restrict__ A, const unsigned short* __restrict__ Bw,
               const float* __restrict__ bias, const float* __restrict__ resid,
               float* __restrict__ Cf, unsigned short* __restrict__ Cb,
               int M, int N, int K) {
  __shared__ unsigned short As[2][128*32];
  __shared__ unsigned short Bs[2][128*32];
  const int tid = threadIdx.x, wave = tid >> 5, lane = tid & 31;
  const int lrow = lane & 15, lhi = lane >> 4;
  const int wm = wave & 3, wn = wave >> 2;
  const int m0 = blockIdx.y * 128, n0 = blockIdx.x * 128;
  v8f acc[2][4];
  #pragma unroll
  for (int a = 0; a < 2; ++a)
    #pragma unroll
    for (int c = 0; c < 4; ++c) acc[a][c] = v8f_zero();

  g_load_tiles(A, Bw, As[0], Bs[0], M, N, K, m0, n0, 0, tid);
  __syncthreads();

  for (int k0 = 0; k0 < K; k0 += 32) {
    const int buf = (k0 >> 5) & 1;
    if (k0 + 32 < K)
      g_load_tiles(A, Bw, As[buf^1], Bs[buf^1], M, N, K, m0, n0, k0 + 32, tid);
    if (k0 + 64 < K) {   // warm L2 two K-steps ahead (global_prefetch_b8)
      int gr = m0 + (tid >> 1); if (gr >= M) gr = M - 1;
      __builtin_prefetch(A + (size_t)gr * K + k0 + 64, 0, 1);
      __builtin_prefetch(Bw + (size_t)(k0 + 64 + (tid & 31)) * N + n0 + ((tid >> 5) << 4), 0, 1);
    }
    FragBF fa[2];
    #pragma unroll
    for (int a = 0; a < 2; ++a) {
      const uint4* asrc = (const uint4*)(&As[buf][(wm*32 + a*16 + lrow)*32]);
      fa[a].u4[0] = asrc[lhi*2]; fa[a].u4[1] = asrc[lhi*2 + 1];
    }
    #pragma unroll
    for (int c = 0; c < 4; ++c) {
      FragBF fb;
      const uint4* bsrc = (const uint4*)(&Bs[buf][(wn*64 + c*16 + lrow)*32]);
      fb.u4[0] = bsrc[lhi*2]; fb.u4[1] = bsrc[lhi*2 + 1];
      #pragma unroll
      for (int a = 0; a < 2; ++a)
        acc[a][c] = __builtin_amdgcn_wmma_f32_16x16x32_bf16(
            false, fa[a].v, false, fb.v, (short)0, acc[a][c], false, false);
    }
    __syncthreads();
  }
  // epilogue: C layout: reg r, lanes0-15 -> (M=r, N=lane), lanes16-31 -> (M=r+8, N=lane-16)
  #pragma unroll
  for (int c = 0; c < 4; ++c) {
    int col = n0 + wn*64 + c*16 + lrow;
    float bv = bias[col];
    #pragma unroll
    for (int a = 0; a < 2; ++a) {
      #pragma unroll
      for (int r = 0; r < 8; ++r) {
        int row = m0 + wm*32 + a*16 + r + 8*lhi;
        if (row < M) {
          float v = acc[a][c][r] + bv;
          if (RESID_) v += resid[(size_t)row * N + col];
          if (GELU_)  v = gelu_exact(v);
          if (OUTBF16_) Cb[(size_t)row * N + col] = f2bf(v);
          else          Cf[(size_t)row * N + col] = v;
        }
      }
    }
  }
}

// ---------------- narrow GEMM for keys (N=64): 128x64 tile, f32 out ----------------
__global__ __launch_bounds__(256)
void gemm_k64(const unsigned short* __restrict__ A, const unsigned short* __restrict__ Bw,
              const float* __restrict__ bias, float* __restrict__ Cf, int M, int N, int K) {
  __shared__ unsigned short As[128*32];
  __shared__ unsigned short Bs[64*32];
  const int tid = threadIdx.x, wave = tid >> 5, lane = tid & 31;
  const int lrow = lane & 15, lhi = lane >> 4;
  const int m0 = blockIdx.y * 128;
  v8f acc[4];
  #pragma unroll
  for (int c = 0; c < 4; ++c) acc[c] = v8f_zero();

  for (int k0 = 0; k0 < K; k0 += 32) {
    __syncthreads();
    {
      int r = tid >> 1, kc = (tid & 1) << 4, gr = m0 + r;
      uint4 d0, d1;
      if (gr < M) {
        const uint4* src = (const uint4*)(A + (size_t)gr * K + k0 + kc);
        d0 = src[0]; d1 = src[1];
      } else { d0 = make_uint4(0,0,0,0); d1 = d0; }
      *(uint4*)&As[r*32 + kc]     = d0;
      *(uint4*)&As[r*32 + kc + 8] = d1;
    }
    {
      int kr = tid >> 3, nc = (tid & 7) << 3;
      uint4 d = *(const uint4*)(Bw + (size_t)(k0 + kr) * N + nc);
      const unsigned short* pv = (const unsigned short*)&d;
      #pragma unroll
      for (int j = 0; j < 8; ++j) Bs[(nc + j)*32 + kr] = pv[j];
    }
    __syncthreads();
    FragBF fa;
    const uint4* asrc = (const uint4*)(As + (wave*16 + lrow)*32);
    fa.u4[0] = asrc[lhi*2]; fa.u4[1] = asrc[lhi*2 + 1];
    #pragma unroll
    for (int c = 0; c < 4; ++c) {
      FragBF fb;
      const uint4* bsrc = (const uint4*)(Bs + (c*16 + lrow)*32);
      fb.u4[0] = bsrc[lhi*2]; fb.u4[1] = bsrc[lhi*2 + 1];
      acc[c] = __builtin_amdgcn_wmma_f32_16x16x32_bf16(
          false, fa.v, false, fb.v, (short)0, acc[c], false, false);
    }
  }
  #pragma unroll
  for (int c = 0; c < 4; ++c) {
    int col = c*16 + lrow;
    float bv = bias[col];
    #pragma unroll
    for (int r = 0; r < 8; ++r) {
      int row = m0 + wave*16 + r + 8*lhi;
      if (row < M) Cf[(size_t)row * N + col] = acc[c][r] + bv;
    }
  }
}

// ---------------- ToMe: normalize 64-d key rows (patches only, in place) ----------------
__global__ void norm_rows64(float* __restrict__ keys) {
  int blk = blockIdx.x;
  int b = blk / NP, p = blk % NP, lane = threadIdx.x;
  float* ptr = keys + ((size_t)b*NN + 1 + p) * HDD;
  float a = ptr[lane], c = ptr[lane + 32];
  float ss = a*a + c*c;
  #pragma unroll
  for (int m = 1; m < 32; m <<= 1) ss += __shfl_xor(ss, m, 32);
  float nrm = fmaxf(sqrtf(ss), 1e-12f);
  ptr[lane] = a / nrm; ptr[lane + 32] = c / nrm;
}

// ---------------- scores = src . dst; row max/argmax ----------------
__global__ __launch_bounds__(256)
void tome_scores(const float* __restrict__ keys, float* __restrict__ nmax, int* __restrict__ nidx) {
  int b = blockIdx.y, s = blockIdx.x * 256 + threadIdx.x;
  if (s >= NSRC) return;
  const float* src = keys + ((size_t)b*NN + 2 + 2*s) * HDD;
  float sv[HDD];
  #pragma unroll 16
  for (int i = 0; i < HDD; ++i) sv[i] = src[i];
  float best = -3.0e38f; int bi = 0;
  for (int j = 0; j < NDST; ++j) {
    const float* d = keys + ((size_t)b*NN + 1 + 2*j) * HDD;
    float acc = 0.f;
    #pragma unroll 16
    for (int i = 0; i < HDD; ++i) acc += sv[i] * d[i];
    if (acc > best) { best = acc; bi = j; }
  }
  nmax[b*NSRC + s] = best; nidx[b*NSRC + s] = bi;
}

// ---------------- per-batch top-k + merge plan ----------------
__global__ void tome_topk(const float* __restrict__ nmax, const int* __restrict__ nidx,
                          int* __restrict__ esrc, int* __restrict__ mdst,
                          float* __restrict__ counts, int* __restrict__ unm) {
  __shared__ float nm[NSRC]; __shared__ int ni[NSRC];
  __shared__ unsigned char sel[NSRC]; __shared__ float cnt[NDST];
  int b = blockIdx.x, tid = threadIdx.x;
  if (tid < NSRC) {
    nm[tid] = nmax[b*NSRC + tid]; ni[tid] = nidx[b*NSRC + tid];
    sel[tid] = 0; cnt[tid] = 1.0f;
  }
  __syncthreads();
  if (tid == 0) {
    for (int i = 0; i < RR; ++i) {
      float best = -3.0e38f; int bi = 0;
      for (int s = 0; s < NSRC; ++s)
        if (!sel[s] && nm[s] > best) { best = nm[s]; bi = s; }
      sel[bi] = 1;
      esrc[b*RR + i] = bi;
      int d = ni[bi]; mdst[b*RR + i] = d; cnt[d] += 1.0f;
    }
    int pidx = 0;
    for (int s = 0; s < NSRC; ++s) if (!sel[s]) unm[b*(NSRC-RR) + pidx++] = s;
  }
  __syncthreads();
  if (tid < NDST) counts[b*NDST + tid] = cnt[tid];
}

// ---------------- assemble merged tokens: [cls | dst | unmerged src] ----------------
__global__ __launch_bounds__(256)
void tome_assemble(const float* __restrict__ x, const int* __restrict__ unm,
                   float* __restrict__ xm) {
  int t = blockIdx.x, b = blockIdx.y, tid = threadIdx.x;
  int srctok;
  if (t == 0)            srctok = 0;
  else if (t <= NDST)    srctok = 1 + 2*(t-1);
  else                   srctok = 2 + 2*unm[b*(NSRC-RR) + (t - NDST - 1)];
  const float* s = x  + ((size_t)b*NN + srctok) * DD;
  float*       d = xm + ((size_t)b*MM + t)     * DD;
  #pragma unroll
  for (int i = 0; i < 4; ++i) d[tid + i*256] = s[tid + i*256];
}

__global__ __launch_bounds__(256)
void tome_scatter(const float* __restrict__ x, const int* __restrict__ esrc,
                  const int* __restrict__ mdst, float* __restrict__ xm) {
  int i = blockIdx.x, b = blockIdx.y, tid = threadIdx.x;
  int s = esrc[b*RR + i], d = mdst[b*RR + i];
  const float* src = x  + ((size_t)b*NN + 2 + 2*s) * DD;
  float*       dst = xm + ((size_t)b*MM + 1 + d)   * DD;
  #pragma unroll
  for (int j = 0; j < 4; ++j) atomicAdd(&dst[tid + j*256], src[tid + j*256]);
}

__global__ __launch_bounds__(256)
void tome_divide(const float* __restrict__ counts, float* __restrict__ xm) {
  int j = blockIdx.x, b = blockIdx.y, tid = threadIdx.x;
  float inv = 1.0f / counts[b*NDST + j];
  float* p = xm + ((size_t)b*MM + 1 + j) * DD;
  #pragma unroll
  for (int i = 0; i < 4; ++i) p[tid + i*256] *= inv;
}

// ---------------- V transpose:  vT[b,h,hd,m(pad 576)] ----------------
__global__ void transpose_v(const unsigned short* __restrict__ qkv2, unsigned short* __restrict__ vT) {
  int idx = blockIdx.x * 256 + threadIdx.x;
  if (idx >= BB*HH*HDD*NP) return;
  int m = idx % NP, rest = idx / NP;
  int hd = rest % HDD, bh = rest / HDD;
  unsigned short val = 0;
  if (m < MM) {
    int b = bh >> 4, h = bh & 15;
    val = qkv2[(size_t)(b*MM + m)*(3*DD) + 2*DD + h*HDD + hd];
  }
  vT[idx] = val;
}

// ---------------- flash attention: 1 wave per 16-query tile, 32-key steps ----------------
__global__ __launch_bounds__(256)
void attn_flash(const unsigned short* __restrict__ qkv, const unsigned short* __restrict__ vT,
                unsigned short* __restrict__ o) {
  __shared__ unsigned short Plds[8][16*32];
  const int bh = blockIdx.x, b = bh >> 4, h = bh & 15;
  const int wave = threadIdx.x >> 5, lane = threadIdx.x & 31;
  const int lrow = lane & 15, lhi = lane >> 4;
  const int qt = blockIdx.y * 8 + wave, qbase = qt * 16;
  const bool qvalid = (qbase < MM);

  FragBF fq0, fq1;
  {
    int q = qbase + lrow; if (q >= MM) q = MM - 1;
    const uint4* qp = (const uint4*)(qkv + (size_t)(b*MM + q)*(3*DD) + h*HDD);
    fq0.u4[0] = qp[lhi*2];     fq0.u4[1] = qp[lhi*2 + 1];      // hd 0..31  A-frag
    fq1.u4[0] = qp[4 + lhi*2]; fq1.u4[1] = qp[4 + lhi*2 + 1];  // hd 32..63 A-frag
  }
  float mrun[8], lrun[8];
  v8f oacc[4];
  #pragma unroll
  for (int r = 0; r < 8; ++r) { mrun[r] = -3.0e38f; lrun[r] = 0.f; }
  #pragma unroll
  for (int c = 0; c < 4; ++c) oacc[c] = v8f_zero();

  const unsigned short* vbase = vT + (size_t)(b*HH + h) * HDD * NP;
  const int nkt = (MM + 31) / 32;                           // 18

  for (int kt = 0; kt < nkt; ++kt) {
    const int kb = kt * 32;
    v8f s0 = v8f_zero(), s1 = v8f_zero();
    #pragma unroll
    for (int sub = 0; sub < 2; ++sub) {
      int kk = kb + sub*16 + lrow; if (kk >= MM) kk = MM - 1;
      const uint4* kp = (const uint4*)(qkv + (size_t)(b*MM + kk)*(3*DD) + DD + h*HDD);
      FragBF fk0, fk1;
      fk0.u4[0] = kp[lhi*2];     fk0.u4[1] = kp[lhi*2 + 1];       // hd 0..31
      fk1.u4[0] = kp[4 + lhi*2]; fk1.u4[1] = kp[4 + lhi*2 + 1];   // hd 32..63
      if (sub == 0) {
        s0 = __builtin_amdgcn_wmma_f32_16x16x32_bf16(false, fq0.v, false, fk0.v, (short)0, s0, false, false);
        s0 = __builtin_amdgcn_wmma_f32_16x16x32_bf16(false, fq1.v, false, fk1.v, (short)0, s0, false, false);
      } else {
        s1 = __builtin_amdgcn_wmma_f32_16x16x32_bf16(false, fq0.v, false, fk0.v, (short)0, s1, false, false);
        s1 = __builtin_amdgcn_wmma_f32_16x16x32_bf16(false, fq1.v, false, fk1.v, (short)0, s1, false, false);
      }
    }
    const bool kv0 = (kb + lrow) < MM, kv1 = (kb + 16 + lrow) < MM;
    #pragma unroll
    for (int r = 0; r < 8; ++r) {
      float a = kv0 ? s0[r] * 0.125f : -1.0e30f;
      float c = kv1 ? s1[r] * 0.125f : -1.0e30f;
      float t  = rmax16(fmaxf(a, c));
      float mn = fmaxf(mrun[r], t);
      float sc = __expf(mrun[r] - mn);
      float p0 = __expf(a - mn), p1 = __expf(c - mn);
      lrun[r] = lrun[r] * sc + rsum16(p0 + p1);
      mrun[r] = mn;
      s0[r] = p0; s1[r] = p1;
      #pragma unroll
      for (int cc = 0; cc < 4; ++cc) oacc[cc][r] *= sc;
    }
    __syncthreads();   // prior-iteration Plds reads complete (uniform loop for all waves)
    #pragma unroll
    for (int r = 0; r < 8; ++r) {
      int row = r + 8*lhi;
      Plds[wave][row*32 + lrow]      = f2bf(s0[r]);
      Plds[wave][row*32 + 16 + lrow] = f2bf(s1[r]);
    }
    __syncthreads();
    FragBF fp;
    const uint4* pp = (const uint4*)(&Plds[wave][lrow*32]);
    fp.u4[0] = pp[lhi*2]; fp.u4[1] = pp[lhi*2 + 1];
    #pragma unroll
    for (int c = 0; c < 4; ++c) {
      int hd = c*16 + lrow;
      const uint4* vp = (const uint4*)(vbase + (size_t)hd * NP + kb + lhi*16);
      FragBF fv; fv.u4[0] = vp[0]; fv.u4[1] = vp[1];
      oacc[c] = __builtin_amdgcn_wmma_f32_16x16x32_bf16(false, fp.v, false, fv.v, (short)0, oacc[c], false, false);
    }
  }
  if (qvalid) {
    #pragma unroll
    for (int c = 0; c < 4; ++c) {
      #pragma unroll
      for (int r = 0; r < 8; ++r) {
        int row = qbase + r + 8*lhi;
        if (row < MM)
          o[(size_t)(b*MM + row)*DD + h*HDD + c*16 + lrow] = f2bf(oacc[c][r] / lrun[r]);
      }
    }
  }
}

// =====================================================================
extern "C" void kernel_launch(void* const* d_in, const int* in_sizes, int n_in,
                              void* d_out, int out_size, void* d_ws, size_t ws_size,
                              hipStream_t stream) {
  (void)in_sizes; (void)n_in; (void)out_size; (void)ws_size;
  const float* x      = (const float*)d_in[0];
  const float* ln1_g  = (const float*)d_in[1];
  const float* ln1_b  = (const float*)d_in[2];
  const float* qkv_w  = (const float*)d_in[3];
  const float* qkv_b  = (const float*)d_in[4];
  const float* proj_w = (const float*)d_in[5];
  const float* proj_b = (const float*)d_in[6];
  const float* ln2_g  = (const float*)d_in[7];
  const float* ln2_b  = (const float*)d_in[8];
  const float* fc1_w  = (const float*)d_in[9];
  const float* fc1_b  = (const float*)d_in[10];
  const float* fc2_w  = (const float*)d_in[11];
  const float* fc2_b  = (const float*)d_in[12];

  char* base = (char*)d_ws;
  size_t off = 0;
  auto take = [&](size_t bytes) -> char* {
    char* r = base + off; off = (off + bytes + 255) & ~(size_t)255; return r;
  };
  unsigned short* wqkv  = (unsigned short*)take((size_t)DD*3*DD*2);
  unsigned short* wproj = (unsigned short*)take((size_t)DD*DD*2);
  unsigned short* wfc1  = (unsigned short*)take((size_t)DD*HIDW*2);
  unsigned short* wfc2  = (unsigned short*)take((size_t)HIDW*DD*2);
  unsigned short* wk    = (unsigned short*)take((size_t)DD*HDD*2);
  float*          bk    = (float*)take(HDD*4);
  unsigned short* xn    = (unsigned short*)take((size_t)TOK1*DD*2);   // reused for all 3 LNs
  float*          keys  = (float*)take((size_t)TOK1*HDD*4);
  float*          nmax  = (float*)take((size_t)BB*NSRC*4);
  int*            nidx  = (int*)take((size_t)BB*NSRC*4);
  int*            esrc  = (int*)take((size_t)BB*RR*4);
  int*            mdst  = (int*)take((size_t)BB*RR*4);
  float*          cnts  = (float*)take((size_t)BB*NDST*4);
  int*            unm   = (int*)take((size_t)BB*(NSRC-RR)*4);
  float*          xm    = (float*)take((size_t)TOK2*DD*4);            // residual stream
  unsigned short* qkv2  = (unsigned short*)take((size_t)TOK2*3*DD*2);
  unsigned short* vT    = (unsigned short*)take((size_t)BB*HH*HDD*NP*2);
  unsigned short* obuf  = (unsigned short*)take((size_t)TOK2*DD*2);
  unsigned short* hbuf  = (unsigned short*)take((size_t)TOK2*HIDW*2);

  // weights -> bf16
  cvt_bf16<<<(DD*3*DD+255)/256, 256, 0, stream>>>(qkv_w, wqkv, DD*3*DD);
  cvt_bf16<<<(DD*DD+255)/256,   256, 0, stream>>>(proj_w, wproj, DD*DD);
  cvt_bf16<<<(DD*HIDW+255)/256, 256, 0, stream>>>(fc1_w, wfc1, DD*HIDW);
  cvt_bf16<<<(HIDW*DD+255)/256, 256, 0, stream>>>(fc2_w, wfc2, HIDW*DD);
  avg_k_weights<<<(DD*HDD+255)/256, 256, 0, stream>>>(qkv_w, qkv_b, wk, bk);

  // pass 1: LN1 + head-averaged keys (1024x64 GEMM)
  layernorm_bf16<<<TOK1, 256, 0, stream>>>(x, ln1_g, ln1_b, xn);
  gemm_k64<<<dim3(1, (TOK1+127)/128), 256, 0, stream>>>(xn, wk, bk, keys, TOK1, HDD, DD);

  // ToMe
  norm_rows64<<<BB*NP, 32, 0, stream>>>(keys);
  tome_scores<<<dim3(2, BB), 256, 0, stream>>>(keys, nmax, nidx);
  tome_topk<<<BB, 288, 0, stream>>>(nmax, nidx, esrc, mdst, cnts, unm);
  tome_assemble<<<dim3(MM, BB), 256, 0, stream>>>(x, unm, xm);
  tome_scatter<<<dim3(RR, BB), 256, 0, stream>>>(x, esrc, mdst, xm);
  tome_divide<<<dim3(NDST, BB), 256, 0, stream>>>(cnts, xm);

  // pass 2: attention
  layernorm_bf16<<<TOK2, 256, 0, stream>>>(xm, ln1_g, ln1_b, xn);
  gemm_bf16<0,0,1><<<dim3(3*DD/128, (TOK2+127)/128), 256, 0, stream>>>(
      xn, wqkv, qkv_b, nullptr, nullptr, qkv2, TOK2, 3*DD, DD);
  transpose_v<<<(BB*HH*HDD*NP+255)/256, 256, 0, stream>>>(qkv2, vT);
  attn_flash<<<dim3(BB*HH, ((MM+15)/16 + 7) / 8), 256, 0, stream>>>(qkv2, vT, obuf);
  gemm_bf16<0,1,0><<<dim3(DD/128, (TOK2+127)/128), 256, 0, stream>>>(
      obuf, wproj, proj_b, xm, xm, nullptr, TOK2, DD, DD);   // in-place residual add

  // MLP
  layernorm_bf16<<<TOK2, 256, 0, stream>>>(xm, ln2_g, ln2_b, xn);
  gemm_bf16<1,0,1><<<dim3(HIDW/128, (TOK2+127)/128), 256, 0, stream>>>(
      xn, wfc1, fc1_b, nullptr, nullptr, hbuf, TOK2, HIDW, DD);
  gemm_bf16<0,1,0><<<dim3(DD/128, (TOK2+127)/128), 256, 0, stream>>>(
      hbuf, wfc2, fc2_b, xm, (float*)d_out, nullptr, TOK2, DD, HIDW);
}
